// JobGraphHGT_86328842649770
// MI455X (gfx1250) — matrix operand order repaired and checked by reading, not compiled, not run on hardware
//
#include <hip/hip_runtime.h>
#include <hip/hip_bf16.h>
#include <math.h>
#include <stdint.h>

typedef __attribute__((ext_vector_type(16))) __bf16 v16bf;
typedef __attribute__((ext_vector_type(8)))  float  v8f;

#define HID   128
#define HEADS 4
#define DH    32

#define GF_RELU_OUT 1
#define GF_GELU_IN  2
#define GF_SKIP     4

__device__ __forceinline__ unsigned short f2bf(float f) {
    unsigned int u = __float_as_uint(f);
    u += 0x7FFFu + ((u >> 16) & 1u);          // round-to-nearest-even
    return (unsigned short)(u >> 16);
}

__device__ __forceinline__ unsigned long long pack4bf(float a, float b, float c, float d) {
    return (unsigned long long)f2bf(a)
         | ((unsigned long long)f2bf(b) << 16)
         | ((unsigned long long)f2bf(c) << 32)
         | ((unsigned long long)f2bf(d) << 48);
}

__device__ __forceinline__ float gelu_tanh(float x) {
    const float c0 = 0.7978845608028654f;
    float x3 = x * x * x;
    return 0.5f * x * (1.0f + tanhf(c0 * (x + 0.044715f * x3)));
}

__device__ __forceinline__ void atomicMaxF(float* a, float v) {
    if (v >= 0.0f) atomicMax((int*)a, __float_as_int(v));
    else           atomicMin((unsigned int*)a, __float_as_uint(v));
}

union FragU { unsigned int u[8]; v16bf v; };

// C[M,N] = epilogue( act_in(A[M,K]) @ W[K,N] + bias ), bf16 WMMA, f32 accumulate.
// Block: 256 threads = 8 waves (4 row x 2 col); block tile 64x64; K-step 64.
// Each wave: one A fragment reused over two 16x16 col tiles, 4 WMMAs / K-step.
template <int FLAGS>
__global__ __launch_bounds__(256) void gemm_bf16_wmma(
    const float* __restrict__ A, int lda,
    const float* __restrict__ W, int ldw,
    const float* __restrict__ bias,
    const float* __restrict__ skipx,
    const float* __restrict__ gatep,
    float* __restrict__ C, int ldc,
    int M, int N, int K)
{
    constexpr bool geluIn  = (FLAGS & GF_GELU_IN)  != 0;
    constexpr bool reluOut = (FLAGS & GF_RELU_OUT) != 0;
    constexpr bool hasSkip = (FLAGS & GF_SKIP)     != 0;

    __shared__ unsigned short As[64][64];   // [m][k]  bf16 bits
    __shared__ unsigned short Bs[64][64];   // [n][k]  bf16 bits (transposed)

    const int tid   = threadIdx.x;
    const int wave  = tid >> 5;
    const int lane  = tid & 31;
    const int wr    = wave >> 1;        // 0..3  (16-row slice)
    const int wc    = wave & 1;         // 0..1  (32-col slice)
    const int l15   = lane & 15;
    const int lhalf = lane >> 4;

    const int row0 = blockIdx.x * 64;
    const int col0 = blockIdx.y * 64;

    v8f acc0 = {0.f, 0.f, 0.f, 0.f, 0.f, 0.f, 0.f, 0.f};
    v8f acc1 = {0.f, 0.f, 0.f, 0.f, 0.f, 0.f, 0.f, 0.f};

    for (int k0 = 0; k0 < K; k0 += 64) {
        // prefetch next A K-panel (global_prefetch_b8)
        if (k0 + 64 < K) {
            int pr = row0 + (tid & 63);
            if (pr < M) __builtin_prefetch(A + (size_t)pr * lda + k0 + 64, 0, 1);
        }
        __syncthreads();
        // stage A: 64 rows x 16 float4-groups, fp32 -> bf16 (optional GELU)
        for (int i = tid; i < 64 * 16; i += 256) {
            int r = i >> 4, g = i & 15;
            int gr = row0 + r;
            int gk = k0 + g * 4;                 // K multiple of 32 -> group never straddles
            float4 v4 = make_float4(0.f, 0.f, 0.f, 0.f);
            if (gr < M && gk < K) v4 = *(const float4*)(A + (size_t)gr * lda + gk);
            if (geluIn) {
                v4.x = gelu_tanh(v4.x); v4.y = gelu_tanh(v4.y);
                v4.z = gelu_tanh(v4.z); v4.w = gelu_tanh(v4.w);
            }
            *(unsigned long long*)&As[r][g * 4] = pack4bf(v4.x, v4.y, v4.z, v4.w);
        }
        // stage B transposed: Bs[n][k] = W[k0+k][col0+n]
        for (int i = tid; i < 64 * 16; i += 256) {
            int n = i >> 4, g = i & 15;
            int gn = col0 + n;
            int gk = k0 + g * 4;
            float x0 = 0.f, x1 = 0.f, x2 = 0.f, x3 = 0.f;
            if (gn < N && gk < K) {
                const float* wp = W + (size_t)gk * ldw + gn;
                x0 = wp[0]; x1 = wp[ldw]; x2 = wp[2 * ldw]; x3 = wp[3 * ldw];
            }
            *(unsigned long long*)&Bs[n][g * 4] = pack4bf(x0, x1, x2, x3);
        }
        __syncthreads();

        const unsigned short* arow = &As[wr * 16 + l15][0];
#pragma unroll
        for (int s = 0; s < 2; ++s) {
            FragU fa;
#pragma unroll
            for (int j = 0; j < 8; ++j) {
                int kkA = s * 32 + lhalf * 16 + 2 * j;
                fa.u[j] = *(const unsigned int*)(arow + kkA);
            }
            FragU fb0, fb1;
            const unsigned short* brow0 = &Bs[wc * 32 + l15][s * 32 + lhalf * 16];
            const unsigned short* brow1 = &Bs[wc * 32 + 16 + l15][s * 32 + lhalf * 16];
#pragma unroll
            for (int j = 0; j < 8; ++j) {
                fb0.u[j] = *(const unsigned int*)(brow0 + 2 * j);
                fb1.u[j] = *(const unsigned int*)(brow1 + 2 * j);
            }
            acc0 = __builtin_amdgcn_wmma_f32_16x16x32_bf16(
                false, fa.v, false, fb0.v, (short)0, acc0, false, false);
            acc1 = __builtin_amdgcn_wmma_f32_16x16x32_bf16(
                false, fa.v, false, fb1.v, (short)0, acc1, false, false);
        }
    }

    float gate = 0.f;
    if (hasSkip) gate = 1.0f / (1.0f + __expf(-gatep[0]));
#pragma unroll
    for (int ct = 0; ct < 2; ++ct) {
        const int n = col0 + wc * 32 + ct * 16 + l15;
        if (n >= N) continue;
#pragma unroll
        for (int r = 0; r < 8; ++r) {
            int m = row0 + wr * 16 + lhalf * 8 + r;
            if (m < M) {
                float v = (ct == 0) ? acc0[r] : acc1[r];
                if (bias) v += bias[n];
                if (reluOut) v = fmaxf(v, 0.f);
                if (hasSkip) v = gate * v + (1.f - gate) * skipx[(size_t)m * ldc + n];
                C[(size_t)m * ldc + n] = v;
            }
        }
    }
}

__global__ void fill_kernel(float* __restrict__ p, float v, size_t n) {
    size_t i = (size_t)blockIdx.x * blockDim.x + threadIdx.x;
    if (i < n) p[i] = v;
}

// per-(edge,head) attention logit + segment max
__global__ void score_kernel(const float* __restrict__ krel,
                             const float* __restrict__ q,
                             const float* __restrict__ pvec,
                             const int* __restrict__ src,
                             const int* __restrict__ dst,
                             float* __restrict__ score,
                             float* __restrict__ mx,
                             size_t dstOff, int E)
{
    int gid = blockIdx.x * blockDim.x + threadIdx.x;
    if (gid >= E * HEADS) return;
    int e = gid >> 2, h = gid & 3;
    int s = src[e], d = dst[e];
    const float4* kr = (const float4*)(krel + (size_t)s * HID + h * DH);
    const float4* qq = (const float4*)(q + (dstOff + (size_t)d) * HID + h * DH);
    float acc = 0.f;
#pragma unroll
    for (int i = 0; i < 8; ++i) {
        float4 a = kr[i], b = qq[i];
        acc += a.x * b.x + a.y * b.y + a.z * b.z + a.w * b.w;
    }
    float sc = acc * pvec[h] * 0.17677669529663687f;   // 1/sqrt(32)
    score[gid] = sc;
    atomicMaxF(&mx[(size_t)d * HEADS + h], sc);
}

// exp(score - max) + segment sum
__global__ void exp_kernel(const int* __restrict__ dst,
                           float* __restrict__ score,
                           const float* __restrict__ mx,
                           float* __restrict__ den, int E)
{
    int gid = blockIdx.x * blockDim.x + threadIdx.x;
    if (gid >= E * HEADS) return;
    int e = gid >> 2, h = gid & 3;
    int d = dst[e];
    float ex = __expf(score[gid] - mx[(size_t)d * HEADS + h]);
    score[gid] = ex;
    atomicAdd(&den[(size_t)d * HEADS + h], ex);
}

// per-(edge,channel) weighted message scatter-add
__global__ void msg_kernel(const float* __restrict__ vrel,
                           const float* __restrict__ exs,
                           const float* __restrict__ den,
                           const int* __restrict__ src,
                           const int* __restrict__ dst,
                           float* __restrict__ agg,
                           size_t dstOff, int E)
{
    size_t gid = (size_t)blockIdx.x * blockDim.x + threadIdx.x;
    if (gid >= (size_t)E * HID) return;
    int e = (int)(gid >> 7);
    int c = (int)(gid & 127);
    int h = c >> 5;
    int s = src[e], d = dst[e];
    float alpha = exs[(size_t)e * HEADS + h] / den[(size_t)d * HEADS + h];
    atomicAdd(&agg[(dstOff + (size_t)d) * HID + c], vrel[(size_t)s * HID + c] * alpha);
}

static void launch_gemm(hipStream_t stream,
                        const float* A, int lda, const float* W, int ldw,
                        const float* bias, const float* skipx, const float* gatep,
                        float* C, int ldc, int M, int N, int K, int flags)
{
    dim3 grid((unsigned)((M + 63) / 64), (unsigned)((N + 63) / 64));
    if (flags == GF_RELU_OUT)
        gemm_bf16_wmma<GF_RELU_OUT><<<grid, 256, 0, stream>>>(
            A, lda, W, ldw, bias, skipx, gatep, C, ldc, M, N, K);
    else if (flags == (GF_GELU_IN | GF_SKIP))
        gemm_bf16_wmma<GF_GELU_IN | GF_SKIP><<<grid, 256, 0, stream>>>(
            A, lda, W, ldw, bias, skipx, gatep, C, ldc, M, N, K);
    else
        gemm_bf16_wmma<0><<<grid, 256, 0, stream>>>(
            A, lda, W, ldw, bias, skipx, gatep, C, ldc, M, N, K);
}

extern "C" void kernel_launch(void* const* d_in, const int* in_sizes, int n_in,
                              void* d_out, int out_size, void* d_ws, size_t ws_size,
                              hipStream_t stream)
{
    (void)in_sizes; (void)n_in; (void)out_size; (void)ws_size;

    const float* x_job     = (const float*)d_in[0];
    const float* x_skill   = (const float*)d_in[1];
    const float* x_company = (const float*)d_in[2];
    const int* ei[4] = {(const int*)d_in[3], (const int*)d_in[4],
                        (const int*)d_in[5], (const int*)d_in[6]};
    const int PB = 7;  // params start; leaves in jax pytree (sorted-key) order

    // type order used here: 0=job, 1=skill, 2=company
    const int    tN[3]    = {100000, 30000, 20000};
    const size_t tOff[3]  = {0, 100000, 130000};
    const int    tFin[3]  = {512, 256, 128};
    const int    tSort[3] = {1, 2, 0};  // sorted position (company,job,skill)
    const float* xin[3]   = {x_job, x_skill, x_company};

    // edge types in reference order: requires, required_by, posted_by, posts
    const int eSrc[4]    = {0, 1, 0, 2};
    const int eDst[4]    = {1, 0, 2, 0};
    const int eE[4]      = {300000, 300000, 100000, 100000};
    const int eRelPos[4] = {2, 3, 1, 0};  // sorted position of rel key

    // workspace layout (floats)
    float* ws = (float*)d_ws;
    const size_t NTOT = 150000;
    float* xd0    = ws;
    float* kbuf   = xd0    + NTOT * HID;
    float* qbuf   = kbuf   + NTOT * HID;
    float* vbuf   = qbuf   + NTOT * HID;
    float* aggb   = vbuf   + NTOT * HID;
    float* xd1    = aggb   + NTOT * HID;
    float* krelb  = xd1    + NTOT * HID;
    float* vrelb  = krelb  + (size_t)100000 * HID;
    float* scoreb = vrelb  + (size_t)100000 * HID;
    float* mxb    = scoreb + (size_t)300000 * HEADS;
    float* denb   = mxb    + (size_t)100000 * HEADS;

    // ---- input projections + ReLU ----
    for (int t = 0; t < 3; ++t) {
        const float* Wp = (const float*)d_in[PB + 2 * tSort[t]];
        const float* bp = (const float*)d_in[PB + 2 * tSort[t] + 1];
        launch_gemm(stream, xin[t], tFin[t], Wp, HID, bp, nullptr, nullptr,
                    xd0 + tOff[t] * HID, HID, tN[t], HID, tFin[t], GF_RELU_OUT);
    }

    float* xd  = xd0;
    float* xdn = xd1;
    for (int l = 0; l < 3; ++l) {
        // layer leaves: k@0, out@6, q@12, rel@18 (4x{a,m,p}), skip@30, v@33  (39 total)
        const int base = PB + 6 + l * 39;

        // ---- K/Q/V projections (bf16 WMMA) ----
        for (int t = 0; t < 3; ++t) {
            const float* Wk = (const float*)d_in[base + 0  + 2 * tSort[t]];
            const float* bk = (const float*)d_in[base + 0  + 2 * tSort[t] + 1];
            const float* Wq = (const float*)d_in[base + 12 + 2 * tSort[t]];
            const float* bq = (const float*)d_in[base + 12 + 2 * tSort[t] + 1];
            const float* Wv = (const float*)d_in[base + 33 + 2 * tSort[t]];
            const float* bv = (const float*)d_in[base + 33 + 2 * tSort[t] + 1];
            launch_gemm(stream, xd + tOff[t] * HID, HID, Wk, HID, bk, nullptr, nullptr,
                        kbuf + tOff[t] * HID, HID, tN[t], HID, HID, 0);
            launch_gemm(stream, xd + tOff[t] * HID, HID, Wq, HID, bq, nullptr, nullptr,
                        qbuf + tOff[t] * HID, HID, tN[t], HID, HID, 0);
            launch_gemm(stream, xd + tOff[t] * HID, HID, Wv, HID, bv, nullptr, nullptr,
                        vbuf + tOff[t] * HID, HID, tN[t], HID, HID, 0);
        }

        // zero the aggregation buffer
        fill_kernel<<<(unsigned)((NTOT * HID + 255) / 256), 256, 0, stream>>>(
            aggb, 0.f, NTOT * HID);

        // ---- per edge type: rel transforms, attention, message passing ----
        for (int et = 0; et < 4; ++et) {
            const int sT = eSrc[et], dT = eDst[et];
            const int E = eE[et];
            const int relBase = base + 18 + 3 * eRelPos[et];
            const float* ra = (const float*)d_in[relBase + 0];  // (H,D,D)
            const float* rm = (const float*)d_in[relBase + 1];  // (H,D,D)
            const float* rp = (const float*)d_in[relBase + 2];  // (H,)

            // krel/vrel: per-head Nsrc x 32 @ 32 x 32 (bf16 WMMA)
            for (int h = 0; h < HEADS; ++h) {
                launch_gemm(stream, kbuf + tOff[sT] * HID + h * DH, HID,
                            ra + h * DH * DH, DH, nullptr, nullptr, nullptr,
                            krelb + h * DH, HID, tN[sT], DH, DH, 0);
                launch_gemm(stream, vbuf + tOff[sT] * HID + h * DH, HID,
                            rm + h * DH * DH, DH, nullptr, nullptr, nullptr,
                            vrelb + h * DH, HID, tN[sT], DH, DH, 0);
            }

            const int nD = tN[dT];
            fill_kernel<<<(nD * HEADS + 255) / 256, 256, 0, stream>>>(
                mxb, -INFINITY, (size_t)nD * HEADS);
            fill_kernel<<<(nD * HEADS + 255) / 256, 256, 0, stream>>>(
                denb, 0.f, (size_t)nD * HEADS);

            const int* srcI = ei[et];
            const int* dstI = ei[et] + E;
            const int nth = E * HEADS;
            score_kernel<<<(nth + 255) / 256, 256, 0, stream>>>(
                krelb, qbuf, rp, srcI, dstI, scoreb, mxb, tOff[dT], E);
            exp_kernel<<<(nth + 255) / 256, 256, 0, stream>>>(
                dstI, scoreb, mxb, denb, E);
            const size_t mth = (size_t)E * HID;
            msg_kernel<<<(unsigned)((mth + 255) / 256), 256, 0, stream>>>(
                vrelb, scoreb, denb, srcI, dstI, aggb, tOff[dT], E);
        }

        // ---- output projection: gelu(agg) @ Wout + b, gated skip with xd ----
        float* dest = (l == 2) ? (float*)d_out : xdn;
        for (int t = 0; t < 3; ++t) {
            const float* Wo = (const float*)d_in[base + 6 + 2 * tSort[t]];
            const float* bo = (const float*)d_in[base + 6 + 2 * tSort[t] + 1];
            const float* gp = (const float*)d_in[base + 30 + tSort[t]];
            launch_gemm(stream, aggb + tOff[t] * HID, HID, Wo, HID, bo,
                        xd + tOff[t] * HID, gp,
                        dest + tOff[t] * HID, HID, tN[t], HID, HID,
                        GF_GELU_IN | GF_SKIP);
        }
        float* tmp = xd; xd = xdn; xdn = tmp;
    }
}
